// bi_LSTM_47218870453093
// MI455X (gfx1250) — matrix operand order repaired and checked by reading, not compiled, not run on hardware
//
#include <hip/hip_runtime.h>
#include <hip/hip_bf16.h>

typedef __attribute__((ext_vector_type(16))) _Float16 v16h;
typedef __attribute__((ext_vector_type(8)))  _Float16 v8h;
typedef __attribute__((ext_vector_type(8)))  float    v8f;

#define B_TOT 4096
#define T_LEN 300
#define EMB   50
#define EPAD  64    // embedding row padded to 64 halves = 128 B
#define V_TOT 20000
#define HID   50
#define G4    200   // 4*H gate width
#define KPAD  128   // padded K for [X | pad | H | pad]
#define NPAD  208   // padded gate width (13 tiles of 16)
#define NTILE 13

// One-time: convert f32 embedding table -> f16, rows padded to 64 halves with
// zeros, so a 16-row gather is exactly 16 x 128 B async-copyable tiles.
__global__ __launch_bounds__(256)
void emb_to_f16_kernel(const float* __restrict__ emb, _Float16* __restrict__ embf16)
{
    const int idx = blockIdx.x * 256 + threadIdx.x;   // over V_TOT*EPAD
    if (idx >= V_TOT * EPAD) return;
    const int row = idx >> 6, col = idx & 63;
    const float v = (col < EMB) ? emb[row * EMB + col] : 0.0f;
    embf16[idx] = (_Float16)v;
}

// One workgroup = 16 batch rows x 1 direction. 128 threads = 4 wave32.
// Weight-stationary: B fragments (f16) live in VGPRs for all 300 steps.
__global__ __launch_bounds__(128)
void bilstm_wmma_kernel(const int*      __restrict__ tokens,
                        const _Float16* __restrict__ embf16,
                        const float* __restrict__ W_f, const float* __restrict__ U_f,
                        const float* __restrict__ b_f,
                        const float* __restrict__ W_b, const float* __restrict__ U_b,
                        const float* __restrict__ b_b,
                        float* __restrict__ hbuf)     // [B][2*HID]
{
    const int tid  = threadIdx.x;
    const int lane = tid & 31;
    const int wave = tid >> 5;
    const int dir  = blockIdx.y;           // 0 = forward, 1 = backward
    const int b0   = blockIdx.x * 16;      // batch tile origin

    const float* Wd = dir ? W_b : W_f;
    const float* Ud = dir ? U_b : U_f;
    const float* bd = dir ? b_b : b_f;

    __shared__ __align__(16) _Float16 Af16[16][KPAD]; // A staging: [X|0|H|0]
    __shared__ float Gbuf[16][NPAD];                  // gate pre-activations
    __shared__ float Cst [16][HID];                   // cell state
    __shared__ int   tokL[16];

    // ---- zero-init A staging (H region + pads) and cell state ----
    for (int idx = tid; idx < 16 * KPAD; idx += 128)
        (&Af16[0][0])[idx] = (_Float16)0.0f;
    for (int idx = tid; idx < 16 * HID; idx += 128)
        (&Cst[0][0])[idx] = 0.0f;

    // ---- build weight-stationary B fragments (f32 global -> f16 VGPR) ----
    // 16-bit B 32x16 layout used here (contiguous K, A matches): lane holds col = nt*16 + lane%16,
    // K = kc*32 + (lane<16 ? 0 : 16) .. +15, consecutive along K.
    const int colLane = lane & 15;
    const int kHalf   = (lane & 16) ? 16 : 0;
    v16h  Bf[4][4];
    float biasr[4];
    #pragma unroll
    for (int i = 0; i < 4; ++i) {
        const int nt  = wave + 4 * i;
        const int col = nt * 16 + colLane;
        biasr[i] = (col < G4) ? bd[col] : 0.0f;
        #pragma unroll
        for (int kc = 0; kc < 4; ++kc) {
            const int kbase = kc * 32 + kHalf;
            #pragma unroll
            for (int j = 0; j < 16; ++j) {
                const int k = kbase + j;
                float v = 0.0f;
                if (col < G4) {
                    if (k < EMB)                       v = Wd[k * G4 + col];
                    else if (k >= 64 && k < 64 + HID)  v = Ud[(k - 64) * G4 + col];
                }
                Bf[i][kc][j] = (_Float16)v;
            }
        }
    }

    // Per-thread constants for the async embedding gather:
    // thread j copies 16 B: row = j/8 of the tile, chunk = j%8 (8 halves).
    const int grow   = tid >> 3;
    const int gchunk = tid & 7;
    const unsigned int ldsDst =
        (unsigned int)(size_t)&Af16[grow][gchunk * 8];

    __syncthreads();

    // ---- sequential scan over time ----
    for (int t = 0; t < T_LEN; ++t) {
        const int tt = dir ? (T_LEN - 1 - t) : t;

        if (tid < 16) {
            tokL[tid] = tokens[(b0 + tid) * T_LEN + tt];
            if (t + 1 < T_LEN) {
                const int tn = dir ? (T_LEN - 2 - t) : (t + 1);
                __builtin_prefetch(&tokens[(b0 + tid) * T_LEN + tn], 0, 1);
            }
        }
        __syncthreads();

        // ---- embedding gather: one async 128-bit global->LDS copy per thread.
        // 128 threads x 16 B == 16 rows x 128 B (f16 row incl. zero padding).
        {
            const unsigned long long gsrc =
                (unsigned long long)(const void*)(embf16 + (size_t)tokL[grow] * EPAD
                                                  + gchunk * 8);
            asm volatile("global_load_async_to_lds_b128 %0, %1, off"
                         :: "v"(ldsDst), "v"(gsrc) : "memory");
            asm volatile("s_wait_asynccnt 0x0" ::: "memory");
        }
        __syncthreads();

        // ---- A fragments: one per K chunk, shared across this wave's N tiles.
        // 16-bit A 16x32 layout (contiguous K, matching B): lanes 0-15 row M, K0-15;
        // lanes 16-31 same rows, K16-31 -> two ds_load_b128 each.
        v16h Af[4];
        const int arow = lane & 15;
        const int ak0  = (lane & 16) ? 16 : 0;
        #pragma unroll
        for (int kc = 0; kc < 4; ++kc) {
            const v8h* p0 = (const v8h*)&Af16[arow][kc * 32 + ak0];
            const v8h* p1 = (const v8h*)&Af16[arow][kc * 32 + ak0 + 8];
            v8h lo = *p0, hi = *p1;
            #pragma unroll
            for (int j = 0; j < 8; ++j) { Af[kc][j] = lo[j]; Af[kc][8 + j] = hi[j]; }
        }

        // ---- G = [X|H] @ [W;U] + b : 16 x v_wmma_f32_16x16x32_f16 per wave ----
        #pragma unroll
        for (int i = 0; i < 4; ++i) {
            const int nt = wave + 4 * i;
            v8f acc;
            #pragma unroll
            for (int r = 0; r < 8; ++r) acc[r] = biasr[i];
            #pragma unroll
            for (int kc = 0; kc < 4; ++kc) {
                acc = __builtin_amdgcn_wmma_f32_16x16x32_f16(
                        false, Af[kc], false, Bf[i][kc], (short)0, acc, false, false);
            }
            if (nt < NTILE) {                       // wave-uniform guard
                const int rhi = (lane & 16) ? 8 : 0;
                #pragma unroll
                for (int r = 0; r < 8; ++r)
                    Gbuf[r + rhi][nt * 16 + colLane] = acc[r];
            }
        }
        __syncthreads();

        // ---- LSTM cell update: i,f,o = relu; cand = tanh; h = o * tanh(c) ----
        for (int idx = tid; idx < 16 * HID; idx += 128) {
            const int row = idx / HID, j = idx - row * HID;
            const float gi = Gbuf[row][j];
            const float gf = Gbuf[row][HID + j];
            const float gc = Gbuf[row][2 * HID + j];
            const float go = Gbuf[row][3 * HID + j];
            const float c  = fmaxf(gf, 0.0f) * Cst[row][j]
                           + fmaxf(gi, 0.0f) * tanhf(gc);
            Cst[row][j] = c;
            const float h = fmaxf(go, 0.0f) * tanhf(c);
            Af16[row][64 + j] = (_Float16)h;        // recurrent input, next step
            if (t == T_LEN - 1)
                hbuf[(b0 + row) * (2 * HID) + dir * HID + j] = h;
        }
        __syncthreads();
    }
}

// Dense(50,relu) -> Dense(1) -> softmax over a single logit == 1.0 exactly.
__global__ __launch_bounds__(128)
void dense_softmax_kernel(const float* __restrict__ hbuf,
                          const float* __restrict__ W1, const float* __restrict__ b1,
                          const float* __restrict__ W2, const float* __restrict__ b2,
                          float* __restrict__ out)
{
    const int b = blockIdx.x * blockDim.x + threadIdx.x;
    if (b >= B_TOT) return;
    float z = b2[0];
    for (int j = 0; j < 50; ++j) {
        float d = b1[j];
        #pragma unroll 4
        for (int k = 0; k < 2 * HID; ++k)
            d = fmaf(hbuf[b * (2 * HID) + k], W1[k * 50 + j], d);
        z = fmaf(fmaxf(d, 0.0f), W2[j], z);
    }
    // softmax([z]) = exp(z - z) / exp(z - z) = 1.0 bit-exactly (matches jax).
    out[b] = 1.0f + 0.0f * z;
}

extern "C" void kernel_launch(void* const* d_in, const int* in_sizes, int n_in,
                              void* d_out, int out_size, void* d_ws, size_t ws_size,
                              hipStream_t stream) {
    (void)in_sizes; (void)n_in; (void)out_size; (void)ws_size;
    const int*   tokens = (const int*)  d_in[0];
    const float* emb    = (const float*)d_in[1];
    const float* W_f    = (const float*)d_in[2];
    const float* U_f    = (const float*)d_in[3];
    const float* b_f    = (const float*)d_in[4];
    const float* W_b    = (const float*)d_in[5];
    const float* U_b    = (const float*)d_in[6];
    const float* b_b    = (const float*)d_in[7];
    const float* W1     = (const float*)d_in[8];
    const float* b1     = (const float*)d_in[9];
    const float* W2     = (const float*)d_in[10];
    const float* b2     = (const float*)d_in[11];

    // Workspace layout: [hbuf: 4096*100 f32 = 1,638,400 B][embf16: 20000*64 f16]
    float*     hbuf   = (float*)d_ws;
    _Float16*  embf16 = (_Float16*)((char*)d_ws + (size_t)B_TOT * 2 * HID * 4);
    float*     out    = (float*)d_out;

    emb_to_f16_kernel<<<dim3((V_TOT * EPAD + 255) / 256, 1, 1), 256, 0, stream>>>(
        emb, embf16);

    dim3 grid1(B_TOT / 16, 2, 1);          // 256 batch tiles x 2 directions
    bilstm_wmma_kernel<<<grid1, 128, 0, stream>>>(tokens, embf16,
                                                  W_f, U_f, b_f,
                                                  W_b, U_b, b_b, hbuf);

    dense_softmax_kernel<<<dim3(B_TOT / 128, 1, 1), 128, 0, stream>>>(
        hbuf, W1, b1, W2, b2, out);
}
